// NodeNet_86474871538493
// MI455X (gfx1250) — matrix operand order repaired and checked by reading, not compiled, not run on hardware
//
#include <hip/hip_runtime.h>
#include <hip/hip_bf16.h>
#include <stdint.h>

typedef __bf16 bf16_t;
typedef __attribute__((ext_vector_type(16))) __bf16 v16bf;
typedef __attribute__((ext_vector_type(8)))  float  v8f;

union FragBF {
  v16bf v;
  float4 f4[2];
};

enum { NODE_F = 128, EDGE_F = 128, GLOB_F = 64, HID = 256, OUT_F = 128 };
enum { K1 = NODE_F + EDGE_F };            // 256, edge-MLP input dim
enum { K2 = NODE_F + HID + GLOB_F };      // 448, node-MLP input dim
enum { STR1 = K1 + 8, STR2 = K2 + 8 };    // LDS row strides (bf16 elems)

#define WMMA_BF16(A, B, C) \
  __builtin_amdgcn_wmma_f32_16x16x32_bf16(false, (A), false, (B), (short)0, (C), false, false)

// ---------------------------------------------------------------- utilities

__global__ void gn_zero_kernel(float* __restrict__ p, long long n) {
  long long i  = (long long)blockIdx.x * blockDim.x + threadIdx.x;
  long long st = (long long)gridDim.x * blockDim.x;
  for (; i < n; i += st) p[i] = 0.0f;
}

// W[k*Nn + n] (f32, k-major) -> Wt[n*K + k] (bf16, transposed, k contiguous)
__global__ void gn_convw_kernel(const float* __restrict__ W, bf16_t* __restrict__ Wt,
                                int K, int Nn) {
  int idx = blockIdx.x * 256 + threadIdx.x;
  if (idx < K * Nn) {
    int n = idx / K;
    int k = idx - n * K;
    Wt[idx] = (bf16_t)W[(long long)k * Nn + n];
  }
}

// ---------------------------------------------------------------- edge model
// 128 threads = 4 waves; 128 edges/block; each wave owns 32 edges (two M-tiles)
// so every B fragment load feeds two WMMAs.
__global__ __launch_bounds__(128) void gn_edge_kernel(
    const float* __restrict__ x, const int* __restrict__ ei,
    const float* __restrict__ ea, const float* __restrict__ b1,
    const bf16_t* __restrict__ W1t, float* __restrict__ summed,
    float* __restrict__ cnt, int E)
{
  __shared__ __align__(16) bf16_t sA[128 * STR1];
  __shared__ int sSrc[128];
  __shared__ int sCol[128];

  const int tid = threadIdx.x;
  const long long e0 = (long long)blockIdx.x * 128;
  const bool full = (e0 + 128 <= (long long)E);   // uniform per block

  {
    long long e = e0 + tid;
    int s = 0, c = 0;
    if (e < E) {
      s = ei[e];
      c = ei[(long long)E + e];
      atomicAdd(&cnt[c], 1.0f);
    }
    sSrc[tid] = s;
    sCol[tid] = c;
  }
  __syncthreads();

  // Stage concat rows into LDS as bf16: thread t covers cols 2t,2t+1 (float2 loads)
  const int c0 = tid * 2;   // 0..254
  for (int r = 0; r < 128; ++r) {
    long long e = e0 + r;
    float2 v = {0.0f, 0.0f};
    if (full || e < E) {
      if (c0 < NODE_F) v = *(const float2*)&x[(long long)sSrc[r] * NODE_F + c0];
      else             v = *(const float2*)&ea[e * EDGE_F + (c0 - NODE_F)];
    }
    union { bf16_t b[2]; unsigned u32; } pk;
    pk.b[0] = (bf16_t)v.x;
    pk.b[1] = (bf16_t)v.y;
    *(unsigned*)&sA[r * STR1 + c0] = pk.u32;
  }
  __syncthreads();

  const int wave = tid >> 5;        // 0..3
  const int lane = tid & 31;
  const int m16  = lane & 15;
  const int g    = lane >> 4;
  const int r0   = wave * 32;       // first of this wave's 32 rows
  const int kb   = g * 8;           // B-fragment K base within a 32-step

  // Preload A fragments for both M-tiles (K = 256 -> 8 k-steps)
  FragBF fa0[K1 / 32], fa1[K1 / 32];
  #pragma unroll
  for (int kk = 0; kk < K1 / 32; ++kk) {
    const bf16_t* p0 = &sA[(r0 + m16) * STR1 + kk * 32 + g * 8];
    fa0[kk].f4[0] = *(const float4*)p0;
    fa0[kk].f4[1] = *(const float4*)(p0 + 16);
    const bf16_t* p1 = p0 + 16 * STR1;
    fa1[kk].f4[0] = *(const float4*)p1;
    fa1[kk].f4[1] = *(const float4*)(p1 + 16);
  }

  for (int nt = 0; nt < HID / 16; ++nt) {
    const int n = nt * 16 + m16;
    const float bias = b1[n];
    v8f acc0 = {bias, bias, bias, bias, bias, bias, bias, bias};
    v8f acc1 = acc0;
    #pragma unroll
    for (int kk = 0; kk < K1 / 32; ++kk) {
      FragBF fb;
      const bf16_t* p = &W1t[(long long)n * K1 + kk * 32 + kb];
      fb.f4[0] = *(const float4*)p;
      fb.f4[1] = *(const float4*)(p + 16);
      acc0 = WMMA_BF16(fa0[kk].v, fb.v, acc0);
      acc1 = WMMA_BF16(fa1[kk].v, fb.v, acc1);
    }
    // ReLU + scatter-add (summed is 51 MB -> atomics resolve in L2)
    if (full) {
      #pragma unroll
      for (int j = 0; j < 8; ++j) {
        const int m = j + 8 * g;
        atomicAdd(&summed[(long long)sCol[r0 + m] * HID + n],      fmaxf(acc0[j], 0.0f));
        atomicAdd(&summed[(long long)sCol[r0 + 16 + m] * HID + n], fmaxf(acc1[j], 0.0f));
      }
    } else {
      #pragma unroll
      for (int j = 0; j < 8; ++j) {
        const int m = j + 8 * g;
        if (e0 + r0 + m < E)
          atomicAdd(&summed[(long long)sCol[r0 + m] * HID + n], fmaxf(acc0[j], 0.0f));
        if (e0 + r0 + 16 + m < E)
          atomicAdd(&summed[(long long)sCol[r0 + 16 + m] * HID + n], fmaxf(acc1[j], 0.0f));
      }
    }
  }
}

// ---------------------------------------------------------------- node model
// h2 = relu([x | summed/cnt | u[batch]] @ W2a + b2a); out = h2 @ W2b + b2b
__global__ __launch_bounds__(256) void gn_node_kernel(
    const float* __restrict__ x, const float* __restrict__ u,
    const int* __restrict__ batch, const float* __restrict__ summed,
    const float* __restrict__ cnt, const float* __restrict__ b2a,
    const float* __restrict__ b2b, const bf16_t* __restrict__ W2at,
    const bf16_t* __restrict__ W2bt, float* __restrict__ out, int N)
{
  __shared__ __align__(16) bf16_t sA[128 * STR2];
  __shared__ float sInv[128];
  __shared__ int   sBat[128];

  const int tid = threadIdx.x;
  const long long i0 = (long long)blockIdx.x * 128;
  const bool full = (i0 + 128 <= (long long)N);   // uniform per block

  if (tid < 128) {
    long long i = i0 + tid;
    float inv = 0.0f; int b = 0;
    if (full || i < N) {
      float c = cnt[i];
      inv = 1.0f / fmaxf(c, 1.0f);
      b = batch[i];
    }
    sInv[tid] = inv;
    sBat[tid] = b;
  }
  __syncthreads();

  for (int r = 0; r < 128; ++r) {
    long long i = i0 + r;
    bool ok = full || (i < N);
    for (int c = tid; c < K2; c += 256) {
      float v = 0.0f;
      if (ok) {
        if (c < NODE_F)            v = x[i * NODE_F + c];
        else if (c < NODE_F + HID) v = summed[i * HID + (c - NODE_F)] * sInv[r];
        else                       v = u[(long long)sBat[r] * GLOB_F + (c - NODE_F - HID)];
      }
      sA[r * STR2 + c] = (bf16_t)v;
    }
  }
  __syncthreads();

  const int wave = tid >> 5, lane = tid & 31;
  const int m16 = lane & 15, g = lane >> 4;
  const int rowBase = wave * 16;
  const int kb = g * 8;

  // ---- GEMM1: [16 x 448] @ [448 x 256] -----------------------------------
  FragBF fa[K2 / 32];  // 14 fragments
  #pragma unroll
  for (int kk = 0; kk < K2 / 32; ++kk) {
    const bf16_t* p = &sA[(rowBase + m16) * STR2 + kk * 32 + g * 8];
    fa[kk].f4[0] = *(const float4*)p;
    fa[kk].f4[1] = *(const float4*)(p + 16);
  }
  for (int nt = 0; nt < HID / 16; ++nt) {
    const int n = nt * 16 + m16;
    const float bias = b2a[n];
    v8f acc = {bias, bias, bias, bias, bias, bias, bias, bias};
    #pragma unroll
    for (int kk = 0; kk < K2 / 32; ++kk) {
      FragBF fb;
      const bf16_t* p = &W2at[(long long)n * K2 + kk * 32 + kb];
      fb.f4[0] = *(const float4*)p;
      fb.f4[1] = *(const float4*)(p + 16);
      acc = WMMA_BF16(fa[kk].v, fb.v, acc);
    }
    // ReLU, write h2 back into this wave's own LDS rows (cols 0..255)
    #pragma unroll
    for (int j = 0; j < 8; ++j) {
      const int m = j + 8 * g;
      sA[(rowBase + m) * STR2 + n] = (bf16_t)fmaxf(acc[j], 0.0f);
    }
  }
  __syncthreads();

  // ---- GEMM2: [16 x 256] @ [256 x 128] -----------------------------------
  FragBF ha[HID / 32];  // 8 fragments
  #pragma unroll
  for (int kk = 0; kk < HID / 32; ++kk) {
    const bf16_t* p = &sA[(rowBase + m16) * STR2 + kk * 32 + g * 8];
    ha[kk].f4[0] = *(const float4*)p;
    ha[kk].f4[1] = *(const float4*)(p + 16);
  }
  for (int ot = 0; ot < OUT_F / 16; ++ot) {
    const int n = ot * 16 + m16;
    const float bias = b2b[n];
    v8f acc = {bias, bias, bias, bias, bias, bias, bias, bias};
    #pragma unroll
    for (int kk = 0; kk < HID / 32; ++kk) {
      FragBF fb;
      const bf16_t* p = &W2bt[(long long)n * HID + kk * 32 + kb];
      fb.f4[0] = *(const float4*)p;
      fb.f4[1] = *(const float4*)(p + 16);
      acc = WMMA_BF16(ha[kk].v, fb.v, acc);
    }
    if (full) {
      #pragma unroll
      for (int j = 0; j < 8; ++j) {
        const int m = j + 8 * g;
        out[(i0 + rowBase + m) * OUT_F + n] = acc[j];
      }
    } else {
      #pragma unroll
      for (int j = 0; j < 8; ++j) {
        const int m = j + 8 * g;
        long long i = i0 + rowBase + m;
        if (i < N) out[i * OUT_F + n] = acc[j];
      }
    }
  }
}

// ---------------------------------------------------------------- launcher

extern "C" void kernel_launch(void* const* d_in, const int* in_sizes, int n_in,
                              void* d_out, int out_size, void* d_ws, size_t ws_size,
                              hipStream_t stream) {
  (void)n_in; (void)out_size; (void)ws_size;

  const float* x   = (const float*)d_in[0];
  const int*   ei  = (const int*)  d_in[1];
  const float* ea  = (const float*)d_in[2];
  const float* u   = (const float*)d_in[3];
  const int*   bat = (const int*)  d_in[4];
  const float* W1  = (const float*)d_in[5];
  const float* b1  = (const float*)d_in[6];
  const float* W2a = (const float*)d_in[7];
  const float* b2a = (const float*)d_in[8];
  const float* W2b = (const float*)d_in[9];
  const float* b2b = (const float*)d_in[10];
  float* out = (float*)d_out;

  const int N = in_sizes[0] / NODE_F;   // 50000
  const int E = in_sizes[2] / EDGE_F;   // 800000

  // Workspace layout: summed [N*HID] f32, cnt [N] f32, then bf16 weights
  float* summed = (float*)d_ws;
  float* cnt    = summed + (size_t)N * HID;
  uintptr_t wp  = ((uintptr_t)(cnt + N) + 255) & ~(uintptr_t)255;
  bf16_t* W1t   = (bf16_t*)wp;
  bf16_t* W2at  = W1t  + (size_t)K1 * HID;
  bf16_t* W2bt  = W2at + (size_t)K2 * HID;

  const long long zn = (long long)N * HID + N;
  gn_zero_kernel<<<2048, 256, 0, stream>>>(summed, zn);
  gn_convw_kernel<<<(K1 * HID    + 255) / 256, 256, 0, stream>>>(W1,  W1t,  K1,  HID);
  gn_convw_kernel<<<(K2 * HID    + 255) / 256, 256, 0, stream>>>(W2a, W2at, K2,  HID);
  gn_convw_kernel<<<(HID * OUT_F + 255) / 256, 256, 0, stream>>>(W2b, W2bt, HID, OUT_F);

  gn_edge_kernel<<<(E + 127) / 128, 128, 0, stream>>>(x, ei, ea, b1, W1t, summed, cnt, E);
  gn_node_kernel<<<(N + 127) / 128, 256, 0, stream>>>(x, u, bat, summed, cnt, b2a, b2b,
                                                      W2at, W2bt, out, N);
}
